// PatchEncoder_14946486190087
// MI455X (gfx1250) — matrix operand order repaired and checked by reading, not compiled, not run on hardware
//
#include <hip/hip_runtime.h>
#include <hip/hip_bf16.h>
#include <math.h>

#define N_NODES 50000
#define N_EDGES 800000
#define N_BATCH 128
#define IN_F    15
#define HID_F   256
#define OUT_F   128
#define LN_EPS  1e-5f

typedef __attribute__((ext_vector_type(16))) __bf16          v16bf;
typedef __attribute__((ext_vector_type(16))) unsigned short  v16u;
typedef __attribute__((ext_vector_type(8)))  unsigned short  v8u;
typedef __attribute__((ext_vector_type(8)))  float           v8f;

// ---- bf16 <-> f32 as raw bit patterns (round-to-nearest-even) ----
__device__ __forceinline__ unsigned short f2bfu(float f) {
    unsigned u = __builtin_bit_cast(unsigned, f);
    u += 0x7FFFu + ((u >> 16) & 1u);
    return (unsigned short)(u >> 16);
}
__device__ __forceinline__ float bfu2f(unsigned short s) {
    unsigned u = ((unsigned)s) << 16;
    return __builtin_bit_cast(float, u);
}

// ---------------- utility kernels ----------------
__global__ void k_zero_f32(float* p, int n) {
    int i = blockIdx.x * 256 + threadIdx.x;
    if (i < n) p[i] = 0.0f;
}

__global__ void k_count(const int* __restrict__ dst, float* cnt, int e) {
    int i = blockIdx.x * 256 + threadIdx.x;
    if (i < e) atomicAdd(&cnt[dst[i]], 1.0f);
}

__global__ void k_inv(const float* __restrict__ cnt, float* inv, int n) {
    int i = blockIdx.x * 256 + threadIdx.x;
    if (i < n) inv[i] = 1.0f / fmaxf(cnt[i], 1.0f);
}

// pad x (N x 15 f32) -> xp (N x 32 bf16, cols 15..31 = 0)
__global__ void k_pad_x(const float* __restrict__ x, unsigned short* xp, int n) {
    int i = blockIdx.x * 256 + threadIdx.x;
    if (i >= n * 32) return;
    int node = i >> 5, k = i & 31;
    xp[i] = (k < IN_F) ? f2bfu(x[node * IN_F + k]) : (unsigned short)0;
}

// mean_bf[i] = bf16(agg[i] * inv[node])   (folds the clip(cnt,1) mean division)
__global__ void k_mean_bf(const float* __restrict__ agg, const float* __restrict__ inv,
                          unsigned short* __restrict__ mb, int n, int ldShift) {
    int i = blockIdx.x * 256 + threadIdx.x;
    if (i >= (n << ldShift)) return;
    int node = i >> ldShift;
    mb[i] = f2bfu(agg[i] * inv[node]);
}

// Pack W (K x outc, f32 row-major) into per-lane B-operand fragments:
// P[((ct*kChunks + kc)*32 + lane)*16 + j] holds column ct*16+(lane&15),
// K = kc*32 + (lane<16 ? 0 : 8) + (j<8 ? j : j+8), zero-padded past K.
__global__ void k_pack_w(const float* __restrict__ W, unsigned short* P,
                         int K, int outc, int kChunks) {
    int total = (outc >> 4) * kChunks * 32;
    int i = blockIdx.x * blockDim.x + threadIdx.x;
    if (i >= total) return;
    int lane = i & 31;
    int kc   = (i >> 5) % kChunks;
    int ct   = (i >> 5) / kChunks;
    int col  = ct * 16 + (lane & 15);
    int kbase = kc * 32 + ((lane < 16) ? 0 : 8);
    unsigned short* d = P + (size_t)i * 16;
#pragma unroll
    for (int j = 0; j < 16; ++j) {
        int k = kbase + (j < 8 ? j : j + 8);
        d[j] = (k < K) ? f2bfu(W[(size_t)k * outc + col]) : (unsigned short)0;
    }
}

// Edge scatter: agg[dst] += h[src]  (bf16 read, fp32 atomic accumulate)
__global__ void k_scatter(const int* __restrict__ src, const int* __restrict__ dst,
                          const unsigned short* __restrict__ h, float* agg,
                          int e, int ld, int chunks) {
    int i = blockIdx.x * 256 + threadIdx.x;
    int tot = e * chunks;
    if (i >= tot) return;
    int edge = i / chunks;
    int c = i - edge * chunks;
    int s = src[edge], d = dst[edge];
    const v8u v = *(const v8u*)(h + (size_t)s * ld + c * 8);
    float* ap = agg + (size_t)d * ld + c * 8;
#pragma unroll
    for (int j = 0; j < 8; ++j) atomicAdd(&ap[j], bfu2f(v[j]));
}

// Dual GEMM on the matrix pipe: C = A0 @ W0 + A1 @ W1   (both A operands bf16)
// One wave -> one 16-row tile x FOUR 16-col tiles: A fragments loaded once,
// reused by 8 WMMAs (4 col tiles x 2 operand pairs) per K-chunk.
__global__ void k_gemm_dual(const unsigned short* __restrict__ A0,
                            const unsigned short* __restrict__ A1,
                            const unsigned short* __restrict__ W0p,
                            const unsigned short* __restrict__ W1p,
                            float* __restrict__ C,
                            int rowTiles, int lda, int ldc, int kChunks) {
    int wave = threadIdx.x >> 5;
    int lane = threadIdx.x & 31;
    int rt = blockIdx.y * (blockDim.x >> 5) + wave;
    if (rt >= rowTiles) return;                 // wave-uniform: EXEC stays all-ones
    int ctBase = blockIdx.x * 4;
    int half = lane >> 4;                       // 0: lanes 0-15, 1: lanes 16-31
    int lrow = lane & 15;
    int row  = rt * 16 + lrow;

    // 16-bit A 16x32 layout: lanes 0-15 hold K j..j+7 at k0 and k0+16; lanes
    // 16-31 the +8 halves -> two 16B loads per operand per chunk.
    const unsigned short* a0 = A0 + (size_t)row * lda + half * 8;
    const unsigned short* a1 = A1 + (size_t)row * lda + half * 8;

    v8f acc[4] = {v8f{}, v8f{}, v8f{}, v8f{}};
    for (int kc = 0; kc < kChunks; ++kc) {
        int k0 = kc * 32;
        v8u p0 = *(const v8u*)(a0 + k0);
        v8u p1 = *(const v8u*)(a0 + k0 + 16);
        v8u q0 = *(const v8u*)(a1 + k0);
        v8u q1 = *(const v8u*)(a1 + k0 + 16);
        v16u ua, uh;
#pragma unroll
        for (int j = 0; j < 8; ++j) {
            ua[j] = p0[j]; ua[8 + j] = p1[j];
            uh[j] = q0[j]; uh[8 + j] = q1[j];
        }
#pragma unroll
        for (int t = 0; t < 4; ++t) {
            int ct = ctBase + t;
            v16u ub0 = *(const v16u*)(W0p + ((size_t)(ct * kChunks + kc) * 32 + lane) * 16);
            v16u ub1 = *(const v16u*)(W1p + ((size_t)(ct * kChunks + kc) * 32 + lane) * 16);
            acc[t] = __builtin_amdgcn_wmma_f32_16x16x32_bf16(
                false, __builtin_bit_cast(v16bf, ua),
                false, __builtin_bit_cast(v16bf, ub0),
                (short)0, acc[t], false, false);
            acc[t] = __builtin_amdgcn_wmma_f32_16x16x32_bf16(
                false, __builtin_bit_cast(v16bf, uh),
                false, __builtin_bit_cast(v16bf, ub1),
                (short)0, acc[t], false, false);
        }
    }
    // C/D layout: VGPR r, lanes 0-15 -> (M=r, N=lane); lanes 16-31 -> (M=8+r, N=lane-16)
#pragma unroll
    for (int t = 0; t < 4; ++t) {
        float* cp = C + (size_t)(rt * 16 + half * 8) * ldc + (ctBase + t) * 16 + lrow;
#pragma unroll
        for (int r = 0; r < 8; ++r) cp[(size_t)r * ldc] = acc[t][r];
    }
}

// Fused bias + LayerNorm + exact GELU, one wave per row, bf16 output.
__global__ void k_ln_gelu(const float* __restrict__ X, const float* __restrict__ bias,
                          const float* __restrict__ g, const float* __restrict__ bt,
                          unsigned short* __restrict__ out, int nRows, int feats) {
    int wave = threadIdx.x >> 5;
    int lane = threadIdx.x & 31;
    int row = blockIdx.x * (blockDim.x >> 5) + wave;
    if (row >= nRows) return;
    int per = feats >> 5;                      // 8 (d=256) or 4 (d=128)
    const float* xp = X + (size_t)row * feats;
    float v[8];
    float s = 0.0f;
    for (int j = 0; j < per; ++j) {
        float t = xp[lane * per + j] + bias[lane * per + j];
        v[j] = t; s += t;
    }
    for (int m = 16; m > 0; m >>= 1) s += __shfl_xor(s, m, 32);
    float mu = s / (float)feats;
    float q = 0.0f;
    for (int j = 0; j < per; ++j) { float d = v[j] - mu; q += d * d; }
    for (int m = 16; m > 0; m >>= 1) q += __shfl_xor(q, m, 32);
    float rstd = rsqrtf(q / (float)feats + LN_EPS);
    unsigned short* op = out + (size_t)row * feats;
    for (int j = 0; j < per; ++j) {
        int idx = lane * per + j;
        float t = (v[j] - mu) * rstd * g[idx] + bt[idx];
        float ge = 0.5f * t * (1.0f + erff(t * 0.70710678118654752f));  // exact GELU
        op[idx] = f2bfu(ge);
    }
}

// ---------------- global mean pool ----------------
__global__ void k_pool_scatter(const int* __restrict__ batch, const unsigned short* __restrict__ h,
                               float* s, int n) {
    int i = blockIdx.x * 256 + threadIdx.x;        // n * (128/8) items
    if (i >= n * 16) return;
    int node = i >> 4, c = i & 15;
    int b = batch[node];
    const v8u v = *(const v8u*)(h + (size_t)node * OUT_F + c * 8);
    float* sp = s + (size_t)b * OUT_F + c * 8;
#pragma unroll
    for (int j = 0; j < 8; ++j) atomicAdd(&sp[j], bfu2f(v[j]));
}

__global__ void k_pool_count(const int* __restrict__ batch, float* c, int n) {
    int i = blockIdx.x * 256 + threadIdx.x;
    if (i < n) atomicAdd(&c[batch[i]], 1.0f);
}

__global__ void k_pool_final(const float* __restrict__ s, const float* __restrict__ c,
                             float* out) {
    int i = blockIdx.x * 256 + threadIdx.x;
    if (i >= N_BATCH * OUT_F) return;
    int b = i / OUT_F;
    out[i] = s[i] / fmaxf(c[b], 1.0f);
}

// ---------------- host orchestration ----------------
static inline int nblk(long long n) { return (int)((n + 255) / 256); }

extern "C" void kernel_launch(void* const* d_in, const int* in_sizes, int n_in,
                              void* d_out, int out_size, void* d_ws, size_t ws_size,
                              hipStream_t stream) {
    (void)in_sizes; (void)n_in; (void)out_size; (void)ws_size;
    const float* x    = (const float*)d_in[0];
    const int*   eidx = (const int*)d_in[1];
    const int*   src  = eidx;
    const int*   dst  = eidx + N_EDGES;
    const int*   batch = (const int*)d_in[2];
    const float* Wl[4] = {(const float*)d_in[3],  (const float*)d_in[8],
                          (const float*)d_in[13], (const float*)d_in[18]};
    const float* Wr[4] = {(const float*)d_in[4],  (const float*)d_in[9],
                          (const float*)d_in[14], (const float*)d_in[19]};
    const float* bb[4] = {(const float*)d_in[5],  (const float*)d_in[10],
                          (const float*)d_in[15], (const float*)d_in[20]};
    const float* gg[4] = {(const float*)d_in[6],  (const float*)d_in[11],
                          (const float*)d_in[16], (const float*)d_in[21]};
    const float* bt[4] = {(const float*)d_in[7],  (const float*)d_in[12],
                          (const float*)d_in[17], (const float*)d_in[22]};
    float* out = (float*)d_out;

    // ---- workspace carve-up ----
    char* ws = (char*)d_ws;
    size_t off = 0;
    auto carve = [&](size_t bytes) -> char* {
        char* p = ws + off;
        off = (off + bytes + 255) & ~(size_t)255;
        return p;
    };
    float*          cnt  = (float*)carve((size_t)N_NODES * 4);
    float*          inv  = (float*)carve((size_t)N_NODES * 4);
    unsigned short* xp   = (unsigned short*)carve((size_t)N_NODES * 32 * 2);
    float*          agg  = (float*)carve((size_t)N_NODES * HID_F * 4);
    unsigned short* mb   = (unsigned short*)carve((size_t)N_NODES * HID_F * 2);
    unsigned short* hA   = (unsigned short*)carve((size_t)N_NODES * HID_F * 2);
    unsigned short* hB   = (unsigned short*)carve((size_t)N_NODES * HID_F * 2);
    float*          gout = (float*)carve((size_t)N_NODES * HID_F * 4);
    // packed weights: [colTiles * kChunks * 32 lanes * 16 bf16]
    unsigned short* Wlp[4]; unsigned short* Wrp[4];
    const int kCh[4]  = {1, 8, 8, 8};
    const int outc[4] = {HID_F, HID_F, HID_F, OUT_F};
    const int Kin[4]  = {IN_F, HID_F, HID_F, HID_F};
    for (int l = 0; l < 4; ++l) {
        size_t n = (size_t)(outc[l] / 16) * kCh[l] * 32 * 16;
        Wlp[l] = (unsigned short*)carve(n * 2);
        Wrp[l] = (unsigned short*)carve(n * 2);
    }
    float* ps = (float*)carve((size_t)N_BATCH * OUT_F * 4);
    float* pc = (float*)carve((size_t)N_BATCH * 4);

    // ---- one-time-per-launch prep ----
    k_zero_f32<<<nblk(N_NODES), 256, 0, stream>>>(cnt, N_NODES);
    k_count<<<nblk(N_EDGES), 256, 0, stream>>>(dst, cnt, N_EDGES);
    k_inv<<<nblk(N_NODES), 256, 0, stream>>>(cnt, inv, N_NODES);
    k_pad_x<<<nblk((long long)N_NODES * 32), 256, 0, stream>>>(x, xp, N_NODES);
    for (int l = 0; l < 4; ++l) {
        int total = (outc[l] / 16) * kCh[l] * 32;
        k_pack_w<<<nblk(total), 256, 0, stream>>>(Wl[l], Wlp[l], Kin[l], outc[l], kCh[l]);
        k_pack_w<<<nblk(total), 256, 0, stream>>>(Wr[l], Wrp[l], Kin[l], outc[l], kCh[l]);
    }

    const int rowTiles = N_NODES / 16;                 // 3125 (exact)
    const int wavesPerBlk = 4;
    dim3 gblk(128, 1, 1);

    auto layer = [&](int l, const unsigned short* hin, unsigned short* hout,
                     int dIn, int ldShift, int scatChunks) {
        k_zero_f32<<<nblk((long long)N_NODES * dIn), 256, 0, stream>>>(agg, N_NODES * dIn);
        k_scatter<<<nblk((long long)N_EDGES * scatChunks), 256, 0, stream>>>(
            src, dst, hin, agg, N_EDGES, dIn, scatChunks);
        k_mean_bf<<<nblk((long long)N_NODES * dIn), 256, 0, stream>>>(
            agg, inv, mb, N_NODES, ldShift);
        dim3 grid((outc[l] / 16) / 4, (rowTiles + wavesPerBlk - 1) / wavesPerBlk, 1);
        k_gemm_dual<<<grid, gblk, 0, stream>>>(mb, hin, Wlp[l], Wrp[l], gout,
                                               rowTiles, dIn, outc[l], kCh[l]);
        k_ln_gelu<<<(N_NODES + wavesPerBlk - 1) / wavesPerBlk, gblk, 0, stream>>>(
            gout, bb[l], gg[l], bt[l], hout, N_NODES, outc[l]);
    };

    // Layer 1: K padded 15->32 (xp/packed-W zero-padded); scatter 16 feats (incl. 1 zero col)
    layer(0, xp, hA, 32, 5, 2);
    layer(1, hA, hB, HID_F, 8, HID_F / 8);
    layer(2, hB, hA, HID_F, 8, HID_F / 8);
    layer(3, hA, hB, HID_F, 8, HID_F / 8);   // hB holds N x 128 bf16

    // ---- global mean pool ----
    k_zero_f32<<<nblk(N_BATCH * OUT_F), 256, 0, stream>>>(ps, N_BATCH * OUT_F);
    k_zero_f32<<<nblk(N_BATCH), 256, 0, stream>>>(pc, N_BATCH);
    k_pool_scatter<<<nblk((long long)N_NODES * 16), 256, 0, stream>>>(batch, hB, ps, N_NODES);
    k_pool_count<<<nblk(N_NODES), 256, 0, stream>>>(batch, pc, N_NODES);
    k_pool_final<<<nblk(N_BATCH * OUT_F), 256, 0, stream>>>(ps, pc, out);
}
